// MultiHeadSelfAttention_39719857553749
// MI455X (gfx1250) — matrix operand order repaired and checked by reading, not compiled, not run on hardware
//
#include <hip/hip_runtime.h>
#include <hip/hip_bf16.h>

// ---------------------------------------------------------------------------
// MI455X (gfx1250) multi-head self-attention, bf16 WMMA, flash-attention core.
// B=2, T=2048, D=1024, H=16, Dh=64.  wave32 only.
// ---------------------------------------------------------------------------

typedef __attribute__((ext_vector_type(16))) __bf16          v16bf;
typedef __attribute__((ext_vector_type(8)))  float           v8f;
typedef __attribute__((ext_vector_type(16))) unsigned short  v16u;
typedef __attribute__((ext_vector_type(8)))  unsigned short  v8u;

__device__ __forceinline__ unsigned short f2bf(float f) {
  // round-to-nearest-even fp32 -> bf16
  unsigned int u = __float_as_uint(f);
  unsigned int r = u + 0x7FFFu + ((u >> 16) & 1u);
  return (unsigned short)(r >> 16);
}

// Build a v16bf A/B fragment from two 16-byte chunks (ds_load_b128 each).
__device__ __forceinline__ v16bf frag_from2(const unsigned short* p0,
                                            const unsigned short* p1) {
  v8u lo = *(const v8u*)p0;
  v8u hi = *(const v8u*)p1;
  v16u t;
#pragma unroll
  for (int i = 0; i < 8; ++i) { t[i] = lo[i]; t[i + 8] = hi[i]; }
  return __builtin_bit_cast(v16bf, t);
}

__device__ __forceinline__ v16bf frag_contig(const unsigned short* p) {
  // 32 contiguous bytes, 32B-aligned at every call site -> two ds_load_b128
  return *(const v16bf*)p;
}

// ---- DPP16 butterfly reductions across the 16-lane row groups of the C
// layout (lanes 0-15 hold rows 0-7, lanes 16-31 rows 8-15; xor masks <16
// never cross the halves).  VALU-only: no LDS traffic, dual-issuable. ----
#define DPP_QUAD_XOR1      0xB1   // quad_perm [1,0,3,2]
#define DPP_QUAD_XOR2      0x4E   // quad_perm [2,3,0,1]
#define DPP_ROW_HALF_MIRR  0x141  // lane ^ 7  within 8
#define DPP_ROW_MIRR       0x140  // lane ^ 15 within 16

template <int CTRL>
__device__ __forceinline__ float dpp_mov_f(float x) {
  int r = __builtin_amdgcn_update_dpp(0, __float_as_int(x), CTRL, 0xF, 0xF, true);
  return __int_as_float(r);
}

__device__ __forceinline__ float row_max16(float v) {
  v = fmaxf(v, dpp_mov_f<DPP_QUAD_XOR1>(v));
  v = fmaxf(v, dpp_mov_f<DPP_QUAD_XOR2>(v));
  v = fmaxf(v, dpp_mov_f<DPP_ROW_HALF_MIRR>(v));
  v = fmaxf(v, dpp_mov_f<DPP_ROW_MIRR>(v));
  return v;
}

__device__ __forceinline__ float row_sum16(float v) {
  v += dpp_mov_f<DPP_QUAD_XOR1>(v);
  v += dpp_mov_f<DPP_QUAD_XOR2>(v);
  v += dpp_mov_f<DPP_ROW_HALF_MIRR>(v);
  v += dpp_mov_f<DPP_ROW_MIRR>(v);
  return v;
}

// ---------------------------------------------------------------------------
// Converters
// ---------------------------------------------------------------------------
__global__ void k_f32_to_bf16(const float* __restrict__ in,
                              unsigned short* __restrict__ out, int n) {
  int i = blockIdx.x * blockDim.x + threadIdx.x;
  if (i < n) out[i] = f2bf(in[i]);
}

// in [R][C] fp32 -> out [C][R] bf16   (weights: B^T so GEMM reads [N][K])
__global__ void k_transpose_f32_to_bf16(const float* __restrict__ in,
                                        unsigned short* __restrict__ out,
                                        int R, int C) {
  int i = blockIdx.x * blockDim.x + threadIdx.x;
  if (i < R * C) {
    int c = i / R, r = i - c * R;            // out is [C][R], flat index i
    out[i] = f2bf(in[(size_t)r * C + c]);
  }
}

// ---------------------------------------------------------------------------
// bf16 WMMA GEMM:  C[M,N] = A[M,K] @ Bt[N,K]^T + bias
//   block tile 128x128 (8 waves, wave tile 32x64), K-step 32 (one WMMA K)
//   MODE 0: QKV epilogue -> scatter bf16 Q(*scale)/K/V as [B,H,T,Dh]
//   MODE 1: fp32 epilogue -> Cout[M,N]
// ---------------------------------------------------------------------------
#define TM 128
#define TN 128
#define TK 32
#define ATTN_SCALE 0.125f   // 64^-0.5

template <int MODE>
__global__ __launch_bounds__(256) void k_gemm_bf16(
    const unsigned short* __restrict__ A,   // [M,K] bf16
    const unsigned short* __restrict__ Bt,  // [N,K] bf16
    const float* __restrict__ bias,         // [N]
    int M, int N, int K,
    unsigned short* __restrict__ Qb, unsigned short* __restrict__ Kb,
    unsigned short* __restrict__ Vb, float* __restrict__ Cout) {
  __shared__ unsigned short As[TM][TK];  // 8 KB
  __shared__ unsigned short Bs[TN][TK];  // 8 KB

  const int tid  = threadIdx.x;
  const int lane = tid & 31;
  const int wave = tid >> 5;
  const int wm   = wave & 3;   // wave M coord (0..3)  -> 32 rows each
  const int wn   = wave >> 2;  // wave N coord (0..1)  -> 64 cols each
  const int mb   = blockIdx.y * TM;
  const int nb   = blockIdx.x * TN;

  v8f acc[2][4] = {};

  // staging assignments (256 threads, 128 rows x 32 k each tile)
  const int srow = tid >> 1;          // 0..127
  const int scol = (tid & 1) * 16;    // 0 / 16

  const int am    = lane & 15;        // A row / B col within 16
  const int abase = (lane >> 4) * 8;  // A K-chunk select per ISA layout
  const int bbase = (lane >> 4) * 16; // contiguous K-half select (unused; B uses abase)

  for (int kb = 0; kb < K; kb += TK) {
    // ---- stage A tile and B' tile (both row-major with contiguous K) ----
    {
      const v8u* ga = (const v8u*)(A + (size_t)(mb + srow) * K + kb + scol);
      *(v8u*)&As[srow][scol]     = ga[0];
      *(v8u*)&As[srow][scol + 8] = ga[1];
      const v8u* gb = (const v8u*)(Bt + (size_t)(nb + srow) * K + kb + scol);
      *(v8u*)&Bs[srow][scol]     = gb[0];
      *(v8u*)&Bs[srow][scol + 8] = gb[1];
      if (kb + TK < K) {
        __builtin_prefetch(A  + (size_t)(mb + srow) * K + kb + TK + scol, 0, 1);
        __builtin_prefetch(Bt + (size_t)(nb + srow) * K + kb + TK + scol, 0, 1);
      }
    }
    __syncthreads();

    // ---- 8 WMMA per wave per K-step (2 M-tiles x 4 N-tiles) ----
#pragma unroll
    for (int mi = 0; mi < 2; ++mi) {
      const unsigned short* ap = &As[wm * 32 + mi * 16 + am][0];
      v16bf afrag = frag_from2(ap + abase, ap + abase + 16);
#pragma unroll
      for (int ni = 0; ni < 4; ++ni) {
        v16bf bfrag = frag_from2(&Bs[wn * 64 + ni * 16 + am][abase], &Bs[wn * 64 + ni * 16 + am][abase + 16]);
        acc[mi][ni] = __builtin_amdgcn_wmma_f32_16x16x32_bf16(
            false, afrag, false, bfrag, (short)0, acc[mi][ni], false, false);
      }
    }
    __syncthreads();
  }

  // ---- epilogue (C layout: VGPR i -> row, lane%16 -> col) ----
  const int rbase = (lane >> 4) * 8;
#pragma unroll
  for (int mi = 0; mi < 2; ++mi) {
#pragma unroll
    for (int ni = 0; ni < 4; ++ni) {
      const int gcol = nb + wn * 64 + ni * 16 + am;
      const float bv = bias[gcol];
#pragma unroll
      for (int i = 0; i < 8; ++i) {
        const int grow = mb + wm * 32 + mi * 16 + rbase + i;
        float v = acc[mi][ni][i] + bv;
        if (MODE == 0) {
          // scatter to Q/K/V [B,H,T,Dh] bf16; Q pre-scaled by 1/sqrt(Dh)
          const int part = gcol >> 10;       // 0=Q 1=K 2=V
          const int d    = gcol & 1023;
          const int h    = d >> 6, dh = d & 63;
          const int bidx = grow >> 11, t = grow & 2047;
          const size_t idx = (((size_t)(bidx * 16 + h)) * 2048 + t) * 64 + dh;
          if (part == 0)      Qb[idx] = f2bf(v * ATTN_SCALE);
          else if (part == 1) Kb[idx] = f2bf(v);
          else                Vb[idx] = f2bf(v);
        } else {
          Cout[(size_t)grow * N + gcol] = v;
        }
      }
    }
  }
}

// ---------------------------------------------------------------------------
// Flash attention: grid (T/64, B*H), 128 threads = 4 waves.
// Each wave owns one 16-row Q tile (A-fragments in registers), streams
// 32-key K/V blocks through LDS, online softmax in fp32 with DPP16
// butterfly reductions (no LDS traffic for the reductions).
// Output Ob is bf16 [B*T, D] row-major (= [B,T,H,Dh]) for the final GEMM.
// ---------------------------------------------------------------------------
__global__ __launch_bounds__(128) void k_flash_attn(
    const unsigned short* __restrict__ Qb, const unsigned short* __restrict__ Kb,
    const unsigned short* __restrict__ Vb, unsigned short* __restrict__ Ob) {
  __shared__ unsigned short Ks[32][64];      // keys row-major      4 KB
  __shared__ unsigned short Vst[64][32];     // V transposed [d][k] 4 KB
  __shared__ unsigned short Ps[4][16][32];   // per-wave P tile     4 KB

  const int tid  = threadIdx.x;
  const int lane = tid & 31;
  const int wave = tid >> 5;
  const int bh   = blockIdx.y;               // b*16 + h
  const int b    = bh >> 4, h = bh & 15;
  const int q0   = blockIdx.x * 64 + wave * 16;
  const size_t head_off = (size_t)bh * 2048 * 64;

  const int am    = lane & 15;
  const int abase = (lane >> 4) * 8;
  const int bbase = (lane >> 4) * 16;

  // Q tile as two A-fragments (head dim split 2x32), loaded once from global
  v16bf qf[2];
#pragma unroll
  for (int c = 0; c < 2; ++c) {
    const unsigned short* qp = Qb + head_off + (size_t)(q0 + am) * 64 + c * 32;
    qf[c] = frag_from2(qp + bbase, qp + bbase + 8);
  }

  v8f   o[4] = {};
  float mrow[8], lrow[8];
#pragma unroll
  for (int i = 0; i < 8; ++i) { mrow[i] = -1e30f; lrow[i] = 0.f; }

  const int krow = tid >> 2;          // 0..31 (key index in block)
  const int kseg = (tid & 3) * 16;    // dim segment

  for (int kb = 0; kb < 2048; kb += 32) {
    // ---- stage K (row-major) and V (transposed) ----
    {
      const v8u* gk = (const v8u*)(Kb + head_off + (size_t)(kb + krow) * 64 + kseg);
      *(v8u*)&Ks[krow][kseg]     = gk[0];
      *(v8u*)&Ks[krow][kseg + 8] = gk[1];
      const v8u* gv = (const v8u*)(Vb + head_off + (size_t)(kb + krow) * 64 + kseg);
      v8u t0 = gv[0], t1 = gv[1];
#pragma unroll
      for (int i = 0; i < 8; ++i) {
        Vst[kseg + i][krow]     = t0[i];
        Vst[kseg + 8 + i][krow] = t1[i];
      }
      if (kb + 32 < 2048) {
        __builtin_prefetch(Kb + head_off + (size_t)(kb + 32 + krow) * 64 + kseg, 0, 1);
        __builtin_prefetch(Vb + head_off + (size_t)(kb + 32 + krow) * 64 + kseg, 0, 1);
      }
    }
    __syncthreads();

    // ---- scores S[16q x 32k] = Qtile @ K^T (4 WMMA) ----
    v8f s[2] = {};
#pragma unroll
    for (int nblk = 0; nblk < 2; ++nblk) {
#pragma unroll
      for (int c = 0; c < 2; ++c) {
        v16bf bf = frag_contig(&Ks[nblk * 16 + am][c * 32 + bbase]);
        s[nblk] = __builtin_amdgcn_wmma_f32_16x16x32_bf16(
            false, qf[c], false, bf, (short)0, s[nblk], false, false);
      }
    }

    // ---- online softmax (per C-layout row; DPP butterfly across 16 lanes) ----
    float p0v[8], p1v[8];
#pragma unroll
    for (int i = 0; i < 8; ++i) {
      float vmax = row_max16(fmaxf(s[0][i], s[1][i]));
      const float mnew  = fmaxf(mrow[i], vmax);
      const float alpha = __expf(mrow[i] - mnew);
      mrow[i] = mnew;
      const float p0 = __expf(s[0][i] - mnew);
      const float p1 = __expf(s[1][i] - mnew);
      p0v[i] = p0; p1v[i] = p1;
      lrow[i] = lrow[i] * alpha + row_sum16(p0 + p1);
#pragma unroll
      for (int nt = 0; nt < 4; ++nt) o[nt][i] *= alpha;
    }

    // ---- C-layout P -> A-layout via per-wave LDS round trip ----
#pragma unroll
    for (int i = 0; i < 8; ++i) {
      const int r = abase + i;
      Ps[wave][r][am]      = f2bf(p0v[i]);
      Ps[wave][r][am + 16] = f2bf(p1v[i]);
    }
    v16bf pf = frag_from2(&Ps[wave][am][bbase], &Ps[wave][am][bbase + 8]);

    // ---- O += P @ V (4 WMMA) ----
#pragma unroll
    for (int nt = 0; nt < 4; ++nt) {
      v16bf vf = frag_contig(&Vst[nt * 16 + am][bbase]);
      o[nt] = __builtin_amdgcn_wmma_f32_16x16x32_bf16(
          false, pf, false, vf, (short)0, o[nt], false, false);
    }
    __syncthreads();
  }

  // ---- normalize and store bf16 to [B*T, D] ----
#pragma unroll
  for (int i = 0; i < 8; ++i) {
    const float rl = 1.0f / lrow[i];
    const int   t  = q0 + abase + i;
    const size_t rowbase = ((size_t)b * 2048 + t) * 1024 + h * 64;
#pragma unroll
    for (int nt = 0; nt < 4; ++nt)
      Ob[rowbase + nt * 16 + am] = f2bf(o[nt][i] * rl);
  }
}

// ---------------------------------------------------------------------------
// Host-side launch
// ---------------------------------------------------------------------------
extern "C" void kernel_launch(void* const* d_in, const int* in_sizes, int n_in,
                              void* d_out, int out_size, void* d_ws, size_t ws_size,
                              hipStream_t stream) {
  const float* x      = (const float*)d_in[0];
  const float* W_qkv  = (const float*)d_in[1];
  const float* b_qkv  = (const float*)d_in[2];
  const float* W_proj = (const float*)d_in[3];
  const float* b_proj = (const float*)d_in[4];
  float* out = (float*)d_out;

  const int B = 2, T = 2048, D = 1024, H = 16;
  const int M = B * T;  // 4096

  // workspace carve-out (~48.5 MB total), 256B aligned slabs
  char* ws = (char*)d_ws;
  size_t off = 0;
  auto alloc = [&](size_t bytes) -> void* {
    void* p = ws + off;
    off = (off + bytes + 255) & ~(size_t)255;
    return p;
  };
  unsigned short* xb     = (unsigned short*)alloc((size_t)M * D * 2);
  unsigned short* wqkvt  = (unsigned short*)alloc((size_t)3 * D * D * 2);
  unsigned short* wprojt = (unsigned short*)alloc((size_t)D * D * 2);
  unsigned short* Qb     = (unsigned short*)alloc((size_t)M * D * 2);
  unsigned short* Kb     = (unsigned short*)alloc((size_t)M * D * 2);
  unsigned short* Vb     = (unsigned short*)alloc((size_t)M * D * 2);
  unsigned short* Ob     = (unsigned short*)alloc((size_t)M * D * 2);
  (void)ws_size; (void)n_in; (void)in_sizes; (void)out_size;

  // 1) precision conversion
  {
    int n = M * D;
    k_f32_to_bf16<<<(n + 255) / 256, 256, 0, stream>>>(x, xb, n);
    n = D * 3 * D;
    k_transpose_f32_to_bf16<<<(n + 255) / 256, 256, 0, stream>>>(W_qkv, wqkvt, D, 3 * D);
    n = D * D;
    k_transpose_f32_to_bf16<<<(n + 255) / 256, 256, 0, stream>>>(W_proj, wprojt, D, D);
  }

  // 2) QKV projection (scatters Q*scale / K / V into [B,H,T,Dh] bf16)
  k_gemm_bf16<0><<<dim3(3 * D / TN, M / TM), 256, 0, stream>>>(
      xb, wqkvt, b_qkv, M, 3 * D, D, Qb, Kb, Vb, nullptr);

  // 3) flash attention -> Ob [B*T, D] bf16
  k_flash_attn<<<dim3(T / 64, B * H), 128, 0, stream>>>(Qb, Kb, Vb, Ob);

  // 4) output projection -> fp32 d_out
  k_gemm_bf16<1><<<dim3(D / TN, M / TM), 256, 0, stream>>>(
      Ob, wprojt, b_proj, M, D, D, nullptr, nullptr, nullptr, out);
}
